// FlexQMixer_20770461843875
// MI455X (gfx1250) — compile-verified, matrix operand order checked by reading, not run on hardware
//
#include <hip/hip_runtime.h>
#include <hip/hip_bf16.h>

// ---------------------------------------------------------------------------
// FlexQMixer on MI455X (gfx1250): fused attention-hypernet per workgroup.
// All matmuls via v_wmma_f32_16x16x32_f16; operand tiles are loaded as
// 2 x b128 per lane (CDNA5 A/B layout = two contiguous 8-half chunks);
// weights/entities pre-converted to f16 once; entity tile staged to LDS
// with global_load_async_to_lds_b128 (ASYNCcnt path).
// ---------------------------------------------------------------------------

typedef __attribute__((ext_vector_type(16))) _Float16 v16h;
typedef __attribute__((ext_vector_type(8)))  _Float16 h8;
typedef __attribute__((ext_vector_type(8)))  float    v8f;
typedef __attribute__((ext_vector_type(4)))  int      v4i;

#define N_AGENTS 16
#define N_HEADS  4
#define HYP      256
#define HD       64
#define EMB      32
#define ENT      128
#define NE       64
#define NITEMS   1024

#if __has_builtin(__builtin_amdgcn_global_load_async_to_lds_b128) && \
    __has_builtin(__builtin_amdgcn_s_wait_asynccnt)
#define HAVE_ASYNC_LDS 1
#else
#define HAVE_ASYNC_LDS 0
#endif

// ---- LDS layout (bytes) ---------------------------------------------------
#define OFF_K      0        // k      : 64x256 f16  (32768)
#define OFF_V      32768    // v      : 64x256 f16  (32768)
#define OFF_X1     65536    // x1     : 64x256 f16  (32768)  [reused: wat, at2]
#define OFF_WAT    65536    //   softmaxed attn weights 4x16x64 f16 (8192)
#define OFF_AT2    73728    //   out-proj result 16x256 f16 (8192)
#define OFF_Q      98304    // q      : 16x256 f16  (8192)   [reused: attn]
#define OFF_E      106496   // ent f16: 64x128     (16384)   [reused: logits f32]
#define OFF_MASK   122880   // 128 ints (em[64], flags)
#define OFF_X3     123392   // 16x32 f32 (2048)
#define LDS_BYTES  125440

// ---- workspace layout (bytes; all 256-aligned) ----------------------------
#define WS_W1M     0u               // 1024*16*32 f32   (2,097,152)
#define WS_B1V     2097152u         // 1024*32 f32      (131,072)
#define WS_WFV     2228224u         // 1024*32 f32      (131,072)
#define WS_VSC     2359296u         // 1024    f32      (4,096)
#define WS_ENT16   2363392u         // 1024*64*128 f16  (16,777,216)
#define WS_WGT16   19140608u        // 4 x 606,208 f16 weight blocks
#define WGT_STRIDE 606208u          // per-hypernet f16 weight block
#define WOFF_FC1   0u               // 256x128 f16 (65,536)
#define WOFF_QKV   65536u           // 768x256 f16 (393,216)
#define WOFF_OUT   458752u          // 256x256 f16 (131,072)
#define WOFF_FC2   589824u          // 32x256  f16 (16,384)

// ---------------------------------------------------------------------------
// Tile loaders. CDNA5 16-bit A (16x32) / row-major-B per-lane layout:
// lane holds row (lane&15); halves 0-7 = K base..base+7, halves 8-15 =
// K base+16..base+23, base = (lane>>4)*8  ->  two contiguous b128 loads.
// ---------------------------------------------------------------------------
__device__ __forceinline__ v16h load_tile16(const _Float16* src, int ld,
                                            int row0, int col0, int lane) {
    const _Float16* p = src + (size_t)(row0 + (lane & 15)) * ld + col0
                            + ((lane >> 4) << 3);
    union { v16h v; h8 h[2]; } u;
    u.h[0] = *(const h8*)(p);
    u.h[1] = *(const h8*)(p + 16);
    return u.v;
}

// Column-access B-tile: B[k][n] = src[(krow0+k)*ld + col0 + n]  (for attn @ v)
__device__ __forceinline__ int kmap(int h, int lane) {
    return ((lane >> 4) << 3) + h + (h & 8);
}
__device__ __forceinline__ v16h load_tile_colB(const _Float16* src, int ld,
                                               int krow0, int col0, int lane) {
    v16h r;
    const int n = col0 + (lane & 15);
#pragma unroll
    for (int h = 0; h < 16; ++h)
        r[h] = src[(size_t)(krow0 + kmap(h, lane)) * ld + n];
    return r;
}

__device__ __forceinline__ v8f wmma_f16(v16h a, v16h b, v8f c) {
    return __builtin_amdgcn_wmma_f32_16x16x32_f16(
        /*neg_a=*/false, a, /*neg_b=*/false, b,
        /*c_mod=*/(short)0, c, /*reuse_a=*/false, /*reuse_b=*/false);
}

// C-tile (16x16 f32) -> f16 LDS. VGPR v: M=v (lanes 0-15), M=v+8 (16-31).
__device__ __forceinline__ void store_tile_f16(_Float16* dst, int ld,
                                               int row0, int col0, v8f c, int lane) {
    const int n  = col0 + (lane & 15);
    const int m0 = row0 + ((lane >> 4) << 3);
#pragma unroll
    for (int v = 0; v < 8; ++v) dst[(size_t)(m0 + v) * ld + n] = (_Float16)c[v];
}

// ---------------------------------------------------------------------------
// fp32 -> f16 bulk conversion (8 elems / thread, b128 stores)
// ---------------------------------------------------------------------------
__global__ __launch_bounds__(256)
void cvt_f16_kernel(const float* __restrict__ src, _Float16* __restrict__ dst,
                    int n8)
{
    const int i = blockIdx.x * 256 + threadIdx.x;
    if (i >= n8) return;
    const float4 a = ((const float4*)src)[i * 2 + 0];
    const float4 b = ((const float4*)src)[i * 2 + 1];
    h8 o;
    o[0] = (_Float16)a.x; o[1] = (_Float16)a.y;
    o[2] = (_Float16)a.z; o[3] = (_Float16)a.w;
    o[4] = (_Float16)b.x; o[5] = (_Float16)b.y;
    o[6] = (_Float16)b.z; o[7] = (_Float16)b.w;
    ((h8*)dst)[i] = o;
}

// ---------------------------------------------------------------------------
// One workgroup == one (batch item, hypernet). mode: 0=matrix 1=vector 2=scalar
// ---------------------------------------------------------------------------
__global__ __launch_bounds__(256)
void hypernet_kernel(const _Float16* __restrict__ ent16,
                     const int*      __restrict__ entity_mask,
                     const _Float16* __restrict__ fc1_w, const float* __restrict__ fc1_b,
                     const _Float16* __restrict__ qkv_w,
                     const _Float16* __restrict__ out_w, const float* __restrict__ out_b,
                     const _Float16* __restrict__ fc2_w, const float* __restrict__ fc2_b,
                     float* __restrict__ out, int mode)
{
    extern __shared__ char smem[];
    _Float16* xk     = (_Float16*)(smem + OFF_K);
    _Float16* xv     = (_Float16*)(smem + OFF_V);
    _Float16* x1     = (_Float16*)(smem + OFF_X1);
    _Float16* wat    = (_Float16*)(smem + OFF_WAT);
    _Float16* at2    = (_Float16*)(smem + OFF_AT2);
    _Float16* qb     = (_Float16*)(smem + OFF_Q);
    _Float16* attn   = (_Float16*)(smem + OFF_Q);    // reuse q region
    _Float16* Eh     = (_Float16*)(smem + OFF_E);
    float*    logits = (float*)   (smem + OFF_E);    // reuse entity region
    int*      mflag  = (int*)     (smem + OFF_MASK);
    float*    x3     = (float*)   (smem + OFF_X3);

    const int item = blockIdx.x;
    const int tid  = threadIdx.x;
    const int wave = tid >> 5;
    const int lane = tid & 31;

    // ---- stage 0: stage f16 entity tile + masks into LDS ------------------
    const _Float16* Eg = ent16 + (size_t)item * NE * ENT;
#if HAVE_ASYNC_LDS
    for (int i = tid; i < (NE * ENT) / 8; i += 256) {
        __builtin_amdgcn_global_load_async_to_lds_b128(
            (__attribute__((address_space(1))) v4i*)(v4i*)(Eg + i * 8),
            (__attribute__((address_space(3))) v4i*)(v4i*)(Eh + i * 8),
            /*offset=*/0, /*cpol=*/0);
    }
    __builtin_amdgcn_s_wait_asynccnt(0);
#else
    for (int i = tid; i < (NE * ENT) / 8; i += 256)
        *(h8*)(Eh + i * 8) = *(const h8*)(Eg + i * 8);
#endif
    if (tid < NE) mflag[tid] = entity_mask[(size_t)item * NE + tid];
    __syncthreads();

    if (tid == 0) {                       // anyEntityValid flag
        int any = 0;
        for (int e = 0; e < NE; ++e) any |= (mflag[e] == 0) ? 1 : 0;
        mflag[64] = any;
    }

    // ---- stage A: x1 = relu(E @ fc1_w^T + fc1_b)  (64x256, K=128) ---------
    for (int t = wave; t < 64; t += 8) {
        const int Mi = t >> 4, Ni = t & 15;
        v8f c = {};
#pragma unroll
        for (int ks = 0; ks < 4; ++ks) {
            v16h a = load_tile16(Eh, ENT, Mi * 16, ks * 32, lane);
            v16h b = load_tile16(fc1_w, ENT, Ni * 16, ks * 32, lane);
            c = wmma_f16(a, b, c);
        }
        const int   n    = Ni * 16 + (lane & 15);
        const float bias = fc1_b[n];
        const int   m0   = Mi * 16 + ((lane >> 4) << 3);
#pragma unroll
        for (int v = 0; v < 8; ++v)
            x1[(size_t)(m0 + v) * HYP + n] = (_Float16)fmaxf(c[v] + bias, 0.0f);
    }
    __syncthreads();

    // ---- stage B: qkv = x1 @ qkv_w^T ; q agents-only (16 rows) ------------
    for (int t = wave; t < 144; t += 8) {
        int Mi, Ni, wrow0;
        _Float16* dst;
        if (t < 16) { Mi = 0; Ni = t; wrow0 = Ni * 16; dst = qb; }
        else {
            int tt = t - 16;
            const int which = tt >> 6; tt &= 63;
            Mi = tt >> 4; Ni = tt & 15;
            wrow0 = (which ? 512 : 256) + Ni * 16;
            dst = which ? xv : xk;
        }
        v8f c = {};
#pragma unroll
        for (int ks = 0; ks < 8; ++ks) {
            v16h a = load_tile16(x1, HYP, Mi * 16, ks * 32, lane);
            v16h b = load_tile16(qkv_w, HYP, wrow0, ks * 32, lane);
            c = wmma_f16(a, b, c);
        }
        store_tile_f16(dst, HYP, Mi * 16, Ni * 16, c, lane);
    }
    __syncthreads();

    // ---- stage C: logits[h][q][e] = (q.k)/8, masked -----------------------
    for (int t = wave; t < 16; t += 8) {
        const int h = t >> 2, Ei = t & 3;
        v8f c = {};
#pragma unroll
        for (int ks = 0; ks < 2; ++ks) {
            v16h a = load_tile16(qb, HYP, 0,       h * HD + ks * 32, lane);
            v16h b = load_tile16(xk, HYP, Ei * 16, h * HD + ks * 32, lane);
            c = wmma_f16(a, b, c);
        }
        const int e  = Ei * 16 + (lane & 15);
        const int em = mflag[e];
        const int q0 = (lane >> 4) << 3;
#pragma unroll
        for (int v = 0; v < 8; ++v) {
            const int qi = q0 + v;
            const int am = mflag[qi];               // agent mask = entity_mask[:16]
            float lg = c[v] * 0.125f;               // 1/sqrt(64)
            if (am != 0 || em != 0) lg = -1e30f;    // keep only am==0 && em==0
            logits[(size_t)(h * 16 + qi) * NE + e] = lg;
        }
    }
    __syncthreads();

    // ---- softmax over e per (h, qi); zero fully-masked rows ---------------
    if (tid < N_HEADS * N_AGENTS) {
        float* row = logits + (size_t)tid * NE;
        float mx = -3.0e38f;
        for (int e = 0; e < NE; ++e) mx = fmaxf(mx, row[e]);
        float s = 0.0f;
        for (int e = 0; e < NE; ++e) { float ex = __expf(row[e] - mx); row[e] = ex; s += ex; }
        const int qi = tid & 15;
        const bool valid = (mflag[qi] == 0) && (mflag[64] != 0);
        const float inv = valid ? (1.0f / s) : 0.0f;
        _Float16* wrow = wat + (size_t)tid * NE;
        for (int e = 0; e < NE; ++e) wrow[e] = (_Float16)(row[e] * inv);
    }
    __syncthreads();

    // ---- stage D: attn = w @ v  (per head 16x64, K=64) --------------------
    for (int t = wave; t < 16; t += 8) {
        const int h = t >> 2, Di = t & 3;
        v8f c = {};
#pragma unroll
        for (int ks = 0; ks < 2; ++ks) {
            v16h a = load_tile16(wat + (size_t)h * 16 * NE, NE, 0, ks * 32, lane);
            v16h b = load_tile_colB(xv, HYP, ks * 32, h * HD + Di * 16, lane);
            c = wmma_f16(a, b, c);
        }
        store_tile_f16(attn, HYP, 0, h * HD + Di * 16, c, lane);
    }
    __syncthreads();

    // ---- stage E: at2 = attn @ out_w^T + out_b, zero masked agent rows ----
    for (int t = wave; t < 16; t += 8) {
        v8f c = {};
#pragma unroll
        for (int ks = 0; ks < 8; ++ks) {
            v16h a = load_tile16(attn, HYP, 0, ks * 32, lane);
            v16h b = load_tile16(out_w, HYP, t * 16, ks * 32, lane);
            c = wmma_f16(a, b, c);
        }
        const int   n    = t * 16 + (lane & 15);
        const float bias = out_b[n];
        const int   m0   = (lane >> 4) << 3;
#pragma unroll
        for (int v = 0; v < 8; ++v) {
            const int m = m0 + v;
            const float x = (mflag[m] != 0) ? 0.0f : (c[v] + bias);
            at2[(size_t)m * HYP + n] = (_Float16)x;
        }
    }
    __syncthreads();

    // ---- stage F: x3 = at2 @ fc2_w^T + fc2_b (16x32), zero masked rows ----
    if (wave < 2) {
        const int Ni = wave;
        v8f c = {};
#pragma unroll
        for (int ks = 0; ks < 8; ++ks) {
            v16h a = load_tile16(at2, HYP, 0, ks * 32, lane);
            v16h b = load_tile16(fc2_w, HYP, Ni * 16, ks * 32, lane);
            c = wmma_f16(a, b, c);
        }
        const int   n    = Ni * 16 + (lane & 15);
        const float bias = fc2_b[n];
        const int   m0   = (lane >> 4) << 3;
#pragma unroll
        for (int v = 0; v < 8; ++v) {
            const int m = m0 + v;
            x3[m * EMB + n] = (mflag[m] != 0) ? 0.0f : (c[v] + bias);
        }
    }
    __syncthreads();

    // ---- mode reduction / writeout ---------------------------------------
    if (mode == 0) {                       // matrix: 16x32 per item
        float* o = out + (size_t)item * N_AGENTS * EMB;
        for (int i = tid; i < N_AGENTS * EMB; i += 256) o[i] = x3[i];
    } else if (mode == 1) {                // vector: mean over agents
        if (tid < EMB) {
            float s = 0.0f;
            for (int a = 0; a < N_AGENTS; ++a) s += x3[a * EMB + tid];
            out[(size_t)item * EMB + tid] = s * (1.0f / 16.0f);
        }
    } else {                               // scalar: mean over 16x32
        if (tid < 32) {
            float s = 0.0f;
            for (int a = 0; a < N_AGENTS; ++a) s += x3[a * EMB + tid];
#pragma unroll
            for (int off = 16; off > 0; off >>= 1) s += __shfl_xor(s, off, 32);
            if (tid == 0) out[item] = s * (1.0f / 512.0f);
        }
    }
}

// ---------------------------------------------------------------------------
// Final mixer: y = elu(qs @ |w1| + b1) . |wf| + v     (one 32-lane wave/item)
// ---------------------------------------------------------------------------
__global__ __launch_bounds__(32)
void mixer_kernel(const float* __restrict__ agent_qs,
                  const float* __restrict__ w1m, const float* __restrict__ b1v,
                  const float* __restrict__ wfv, const float* __restrict__ vs,
                  float* __restrict__ out)
{
    const int item = blockIdx.x;
    const int e    = threadIdx.x;          // 0..31 == EMB
    const float* qs = agent_qs + (size_t)item * N_AGENTS;
    const float* W  = w1m + (size_t)item * N_AGENTS * EMB;

    float acc = 0.0f;
#pragma unroll
    for (int a = 0; a < N_AGENTS; ++a) acc += qs[a] * fabsf(W[a * EMB + e]);
    acc += b1v[(size_t)item * EMB + e];
    const float h = (acc > 0.0f) ? acc : expm1f(acc);          // elu, alpha=1
    float p = h * fabsf(wfv[(size_t)item * EMB + e]);
#pragma unroll
    for (int off = 16; off > 0; off >>= 1) p += __shfl_xor(p, off, 32);
    if (e == 0) out[item] = p + vs[item];
}

// ---------------------------------------------------------------------------
extern "C" void kernel_launch(void* const* d_in, const int* in_sizes, int n_in,
                              void* d_out, int out_size, void* d_ws, size_t ws_size,
                              hipStream_t stream)
{
    (void)in_sizes; (void)n_in; (void)out_size; (void)ws_size;

    const float* agent_qs = (const float*)d_in[0];
    const float* entities = (const float*)d_in[1];
    const int*   emask    = (const int*)d_in[2];

    char* ws = (char*)d_ws;
    float*     w1m   = (float*)(ws + WS_W1M);
    float*     b1v   = (float*)(ws + WS_B1V);
    float*     wfv   = (float*)(ws + WS_WFV);
    float*     vsc   = (float*)(ws + WS_VSC);
    _Float16*  ent16 = (_Float16*)(ws + WS_ENT16);

    // ---- prep: fp32 -> f16 for entities and all weight matrices -----------
    {
        const int n8 = (NITEMS * NE * ENT) / 8;
        cvt_f16_kernel<<<(n8 + 255) / 256, 256, 0, stream>>>(entities, ent16, n8);
    }
    static const unsigned wofs[4]  = { WOFF_FC1, WOFF_QKV, WOFF_OUT, WOFF_FC2 };
    static const int      welem[4] = { 256 * 128, 768 * 256, 256 * 256, 32 * 256 };
    static const int      wsrc[4]  = { 0, 2, 3, 5 };   // index within 7-ptr group
    for (int hn = 0; hn < 4; ++hn) {
        _Float16* wbase = (_Float16*)(ws + WS_WGT16 + (size_t)hn * WGT_STRIDE);
        for (int m = 0; m < 4; ++m) {
            const int n8 = welem[m] / 8;
            cvt_f16_kernel<<<(n8 + 255) / 256, 256, 0, stream>>>(
                (const float*)d_in[3 + hn * 7 + wsrc[m]],
                (_Float16*)((char*)wbase + wofs[m]), n8);
        }
    }

    // ---- 4 hypernet passes -------------------------------------------------
    float* outs[4]     = { w1m, b1v, wfv, vsc };
    const int modes[4] = { 0, 1, 1, 2 };
    for (int hn = 0; hn < 4; ++hn) {
        const int base = 3 + hn * 7;
        const char* wb = ws + WS_WGT16 + (size_t)hn * WGT_STRIDE;
        hypernet_kernel<<<NITEMS, 256, LDS_BYTES, stream>>>(
            ent16, emask,
            (const _Float16*)(wb + WOFF_FC1),
            (const float*)d_in[base + 1],            // fc1_b
            (const _Float16*)(wb + WOFF_QKV),
            (const _Float16*)(wb + WOFF_OUT),
            (const float*)d_in[base + 4],            // out_b
            (const _Float16*)(wb + WOFF_FC2),
            (const float*)d_in[base + 6],            // fc2_b
            outs[hn], modes[hn]);
    }

    mixer_kernel<<<NITEMS, 32, 0, stream>>>(agent_qs, w1m, b1v, wfv, vsc,
                                            (float*)d_out);
}
